// HNNModel_17145509445712
// MI455X (gfx1250) — matrix operand order repaired and checked by reading, not compiled, not run on hardware
//
#include <hip/hip_runtime.h>
#include <hip/hip_bf16.h>

// ---------------------------------------------------------------------------
// HNN (hybrid SNN/RNN) on gfx1250: bf16 WMMA 16x16x32 + async global->LDS A
// staging. B=4096, T=28, hidden=512. 28 sequential steps, each step =
// layer1 kernel + layer2 kernel. Wave tile = 32M x 32N (8 f32 accumulators).
// ---------------------------------------------------------------------------

typedef __attribute__((ext_vector_type(16))) __bf16 v16bf;
typedef __attribute__((ext_vector_type(8)))  float  v8f;
typedef __attribute__((ext_vector_type(4)))  int    v4i;

#define BB 4096
#define HB 512
#define TT 28
#define LDST 520   // LDS row stride (bf16) for 512-wide tiles: +8 pad kills bank conflicts
#define XST  40    // LDS row stride for 32-wide x tiles

#define AS1 __attribute__((address_space(1)))
#define AS3 __attribute__((address_space(3)))

#if __has_builtin(__builtin_amdgcn_global_load_async_to_lds_b128)
#define HNN_ASYNC 1
#endif

// 16-byte global -> LDS copy (async path tracked by ASYNCcnt when available)
__device__ __forceinline__ void cp16(const __bf16* __restrict__ g,
                                     __bf16* __restrict__ l) {
#ifdef HNN_ASYNC
  __builtin_amdgcn_global_load_async_to_lds_b128(
      (AS1 v4i*)(g), (AS3 v4i*)(l), 0, 0);
#else
  *(uint4*)l = *(const uint4*)g;
#endif
}

__device__ __forceinline__ void stage_fence() {
#ifdef HNN_ASYNC
#if __has_builtin(__builtin_amdgcn_s_wait_asynccnt)
  __builtin_amdgcn_s_wait_asynccnt(0);
#else
  asm volatile("s_wait_asynccnt 0" ::: "memory");
#endif
#endif
  __syncthreads();
}

// ---- fragment loaders (ISA 7.12.2 layouts, wave32) ------------------------

// A-matrix 16x32 bf16 fragment from an LDS (or any) row-major tile.
// lane l: M = l&15, h = l>>4; elems 0-7 <- K = h*8+0..7, elems 8-15 <- K = 16+h*8+0..7
__device__ __forceinline__ v16bf lds_a16x32(const __bf16* __restrict__ A, int ldr,
                                            int mrow0, int kt, int lane) {
  const int m = lane & 15, h = lane >> 4;
  const __bf16* p = A + (size_t)(mrow0 + m) * ldr + kt * 32 + h * 8;
  union { uint4 u[2]; v16bf v; } t;
  t.u[0] = *(const uint4*)(p);
  t.u[1] = *(const uint4*)(p + 16);
  return t.v;
}

// B-matrix 32x16 bf16 from pre-packed weights: [kt][nt][lane][16] contiguous,
// lane l holds N = l&15, K = (l>>4)*8 + 0..7 then 16 + (l>>4)*8 + 0..7 -> one aligned 32B load.
__device__ __forceinline__ v16bf load_b32x16(const __bf16* __restrict__ Wp,
                                             int ntiles, int kt, int nt, int lane) {
  return *(const v16bf*)(Wp + ((((size_t)kt * ntiles + nt) * 32 + lane) << 4));
}

__device__ __forceinline__ v8f wmma_bf16(v16bf a, v16bf b, v8f c) {
  return __builtin_amdgcn_wmma_f32_16x16x32_bf16(
      /*neg_a=*/false, a, /*neg_b=*/false, b,
      /*c_mod=*/(short)0, c, /*reuse_a=*/false, /*reuse_b=*/false);
}

__device__ __forceinline__ float sigmoidf_(float x) {
  return 1.f / (1.f + __expf(-x));
}

// ---- prep kernels ---------------------------------------------------------

// x [B, 784] -> xt [T][B][32] bf16 with xt[t][b][k] = x[b, k*28 + t], K padded to 32.
__global__ void hnn_pack_x(const float* __restrict__ x, __bf16* __restrict__ xt) {
  const size_t idx = (size_t)blockIdx.x * blockDim.x + threadIdx.x;
  if (idx >= (size_t)TT * BB * 32) return;
  const int k = (int)(idx & 31);
  const int b = (int)((idx >> 5) & (BB - 1));
  const int t = (int)(idx >> 17); // 5 + 12
  const float v = (k < 28) ? x[(size_t)b * 784 + (size_t)k * 28 + t] : 0.f;
  xt[idx] = (__bf16)v;
}

// W [K, N] row-major f32 -> packed bf16 [Kp/32][N/16][32 lanes][16], zero-padded K.
__global__ void hnn_pack_w(const float* __restrict__ W, __bf16* __restrict__ Wp,
                           int K, int Kp, int N) {
  const int idx = blockIdx.x * blockDim.x + threadIdx.x;
  if (idx >= Kp * N) return;
  const int e      = idx & 15;
  const int l      = (idx >> 4) & 31;
  const int rest   = idx >> 9;
  const int ntiles = N >> 4;
  const int nt = rest % ntiles;
  const int kt = rest / ntiles;
  const int k = kt * 32 + (e >> 3) * 16 + (l >> 4) * 8 + (e & 7);
  const int n = nt * 16 + (l & 15);
  const float v = (k < K) ? W[(size_t)k * N + n] : 0.f;
  Wp[idx] = (__bf16)v;
}

// ---- step kernels ---------------------------------------------------------
// Grid: 256 blocks x 256 threads (8 waves).
// mt2 = blockIdx>>1 (32-row strip), ns2 = (blockIdx&1)*8 + wave (32-col strip).
// Block stages its 32 A-rows into LDS once (async), waves tile 32M x 32N.

__global__ __launch_bounds__(256) void hnn_layer1(
    const __bf16* __restrict__ xt,    // [BB][32]  (this step's column of x)
    const __bf16* __restrict__ w_s1,  // snn_fc1 packed
    const __bf16* __restrict__ w_r1,  // rnn_fc1 packed
    const __bf16* __restrict__ w_v1,  // rnn_fv1 packed [16][32][.][.]
    const __bf16* __restrict__ y_in,  // h1_y prev [BB][HB]
    __bf16*       __restrict__ y_out, // h1_y next
    __bf16*       __restrict__ spk,   // h1 spikes (prev in / new out)
    float*        __restrict__ mem,   // h1_mem
    float*        __restrict__ y_f32, // fp32 h1_y (final step only)
    int store_f32)
{
  __shared__ __bf16 sX[32 * XST];
  __shared__ __bf16 sY[32 * LDST];

  const int tid  = threadIdx.x;
  const int lane = tid & 31;
  const int wid  = tid >> 5;
  const int r0   = (blockIdx.x >> 1) * 32;

  // stage x rows (32 x 32 bf16 -> 128 x 16B chunks)
  for (int c = tid; c < 128; c += 256) {
    const int row = c >> 2, cc = c & 3;
    cp16(xt + (size_t)(r0 + row) * 32 + cc * 8, sX + row * XST + cc * 8);
  }
  // stage h1_y rows (32 x 256 -> 1024 chunks)
  for (int c = tid; c < 1024; c += 256) {
    const int row = c >> 5, cc = c & 31;
    cp16(y_in + (size_t)(r0 + row) * HB + cc * 8, sY + row * LDST + cc * 8);
  }
  stage_fence();

  const int ns2 = ((blockIdx.x & 1) << 3) + wid;  // 0..15
  const int nt0 = ns2 * 2;

  v8f acc_s[2][2] = {}, acc_r[2][2] = {};

  // input projections, K = 32 (padded 28)
  {
    const v16bf a0 = lds_a16x32(sX, XST, 0, 0, lane);
    const v16bf a1 = lds_a16x32(sX, XST, 16, 0, lane);
#pragma unroll
    for (int ni = 0; ni < 2; ++ni) {
      const v16bf bs = load_b32x16(w_s1, 32, 0, nt0 + ni, lane);
      const v16bf br = load_b32x16(w_r1, 32, 0, nt0 + ni, lane);
      acc_s[0][ni] = wmma_bf16(a0, bs, acc_s[0][ni]);
      acc_s[1][ni] = wmma_bf16(a1, bs, acc_s[1][ni]);
      acc_r[0][ni] = wmma_bf16(a0, br, acc_r[0][ni]);
      acc_r[1][ni] = wmma_bf16(a1, br, acc_r[1][ni]);
    }
  }
  // recurrent h1_y @ rnn_fv1, K = 512
#pragma unroll 4
  for (int kt = 0; kt < 16; ++kt) {
    const v16bf a0 = lds_a16x32(sY, LDST, 0, kt, lane);
    const v16bf a1 = lds_a16x32(sY, LDST, 16, kt, lane);
#pragma unroll
    for (int ni = 0; ni < 2; ++ni) {
      const v16bf bv = load_b32x16(w_v1, 32, kt, nt0 + ni, lane);
      acc_r[0][ni] = wmma_bf16(a0, bv, acc_r[0][ni]);
      acc_r[1][ni] = wmma_bf16(a1, bv, acc_r[1][ni]);
    }
  }

  // epilogue: LIF membrane update + spike + sigmoid on C/D register layout
  const int half = lane >> 4, li = lane & 15;
#pragma unroll
  for (int mi = 0; mi < 2; ++mi) {
#pragma unroll
    for (int ni = 0; ni < 2; ++ni) {
      const int c = (nt0 + ni) * 16 + li;
#pragma unroll
      for (int i = 0; i < 8; ++i) {
        const int r = r0 + mi * 16 + half * 8 + i;
        const size_t idx = (size_t)r * HB + c;
        const float so = (float)spk[idx];
        const float m  = mem[idx] * (1.f - so) * 0.6f + acc_s[mi][ni][i];
        const float sn = (m - 0.6f) > 0.f ? 1.f : 0.f;
        mem[idx] = m;
        spk[idx] = (__bf16)sn;
        const float y = sigmoidf_(acc_r[mi][ni][i]);
        y_out[idx] = (__bf16)y;
        if (store_f32) y_f32[idx] = y;
      }
    }
  }
}

__global__ __launch_bounds__(256) void hnn_layer2(
    const __bf16* __restrict__ spk1,  // new h1 spikes (o1 cols 0..511)
    const __bf16* __restrict__ y1,    // new h1_y      (o1 cols 512..1023)
    const __bf16* __restrict__ w_s2,  // snn_fc2 packed [32][32][.][.]
    const __bf16* __restrict__ w_r2,  // rnn_fc2 packed
    const __bf16* __restrict__ w_v2,  // rnn_fv2 packed [16][32][.][.]
    const __bf16* __restrict__ y_in,  // h2_y prev
    __bf16*       __restrict__ y_out, // h2_y next
    float*        __restrict__ spk2,  // h2 spikes fp32
    float*        __restrict__ mem,   // h2_mem
    float*        __restrict__ ssum,  // h2 spike sum
    float*        __restrict__ y_f32, // fp32 h2_y (final step only)
    int store_f32)
{
  __shared__ __bf16 sS [32 * LDST];
  __shared__ __bf16 sY1[32 * LDST];
  __shared__ __bf16 sY2[32 * LDST];

  const int tid  = threadIdx.x;
  const int lane = tid & 31;
  const int wid  = tid >> 5;
  const int r0   = (blockIdx.x >> 1) * 32;

  for (int c = tid; c < 1024; c += 256) {
    const int row = c >> 5, cc = c & 31;
    const size_t go = (size_t)(r0 + row) * HB + cc * 8;
    const int    lo = row * LDST + cc * 8;
    cp16(spk1 + go, sS  + lo);
    cp16(y1   + go, sY1 + lo);
    cp16(y_in + go, sY2 + lo);
  }
  stage_fence();

  const int ns2 = ((blockIdx.x & 1) << 3) + wid;
  const int nt0 = ns2 * 2;

  v8f acc_s[2][2] = {}, acc_r[2][2] = {};

  // o1 left half (spikes): weight k-tiles 0..15
#pragma unroll 2
  for (int kt = 0; kt < 16; ++kt) {
    const v16bf a0 = lds_a16x32(sS, LDST, 0, kt, lane);
    const v16bf a1 = lds_a16x32(sS, LDST, 16, kt, lane);
#pragma unroll
    for (int ni = 0; ni < 2; ++ni) {
      const v16bf bs = load_b32x16(w_s2, 32, kt, nt0 + ni, lane);
      const v16bf br = load_b32x16(w_r2, 32, kt, nt0 + ni, lane);
      acc_s[0][ni] = wmma_bf16(a0, bs, acc_s[0][ni]);
      acc_s[1][ni] = wmma_bf16(a1, bs, acc_s[1][ni]);
      acc_r[0][ni] = wmma_bf16(a0, br, acc_r[0][ni]);
      acc_r[1][ni] = wmma_bf16(a1, br, acc_r[1][ni]);
    }
  }
  // o1 right half (h1_y): weight k-tiles 16..31
#pragma unroll 2
  for (int kt = 0; kt < 16; ++kt) {
    const v16bf a0 = lds_a16x32(sY1, LDST, 0, kt, lane);
    const v16bf a1 = lds_a16x32(sY1, LDST, 16, kt, lane);
#pragma unroll
    for (int ni = 0; ni < 2; ++ni) {
      const v16bf bs = load_b32x16(w_s2, 32, kt + 16, nt0 + ni, lane);
      const v16bf br = load_b32x16(w_r2, 32, kt + 16, nt0 + ni, lane);
      acc_s[0][ni] = wmma_bf16(a0, bs, acc_s[0][ni]);
      acc_s[1][ni] = wmma_bf16(a1, bs, acc_s[1][ni]);
      acc_r[0][ni] = wmma_bf16(a0, br, acc_r[0][ni]);
      acc_r[1][ni] = wmma_bf16(a1, br, acc_r[1][ni]);
    }
  }
  // recurrent h2_y @ rnn_fv2
#pragma unroll 2
  for (int kt = 0; kt < 16; ++kt) {
    const v16bf a0 = lds_a16x32(sY2, LDST, 0, kt, lane);
    const v16bf a1 = lds_a16x32(sY2, LDST, 16, kt, lane);
#pragma unroll
    for (int ni = 0; ni < 2; ++ni) {
      const v16bf bv = load_b32x16(w_v2, 32, kt, nt0 + ni, lane);
      acc_r[0][ni] = wmma_bf16(a0, bv, acc_r[0][ni]);
      acc_r[1][ni] = wmma_bf16(a1, bv, acc_r[1][ni]);
    }
  }

  const int half = lane >> 4, li = lane & 15;
#pragma unroll
  for (int mi = 0; mi < 2; ++mi) {
#pragma unroll
    for (int ni = 0; ni < 2; ++ni) {
      const int c = (nt0 + ni) * 16 + li;
#pragma unroll
      for (int i = 0; i < 8; ++i) {
        const int r = r0 + mi * 16 + half * 8 + i;
        const size_t idx = (size_t)r * HB + c;
        const float so = spk2[idx];
        const float m  = mem[idx] * (1.f - so) * 0.6f + acc_s[mi][ni][i];
        const float sn = (m - 0.6f) > 0.f ? 1.f : 0.f;
        mem[idx]  = m;
        spk2[idx] = sn;
        ssum[idx] += sn;
        const float y = sigmoidf_(acc_r[mi][ni][i]);
        y_out[idx] = (__bf16)y;
        if (store_f32) y_f32[idx] = y;
      }
    }
  }
}

// ---- classifier head: logits = concat(h2_sum/28, h2_y) @ fc_w.T + fc_b ----
__global__ void hnn_head(const float* __restrict__ ssum,
                         const float* __restrict__ y2_f32,
                         const float* __restrict__ fc_w,   // [10][1024]
                         const float* __restrict__ fc_b,   // [10]
                         float* __restrict__ logits)       // [BB][10]
{
  const int idx = blockIdx.x * blockDim.x + threadIdx.x;
  if (idx >= BB * 10) return;
  const int b = idx / 10, c = idx % 10;
  const float* w = fc_w + (size_t)c * 1024;
  float s = fc_b[c];
  const float inv_t = 1.f / 28.f;
#pragma unroll 4
  for (int k = 0; k < HB; ++k) s += ssum[(size_t)b * HB + k] * inv_t * w[k];
#pragma unroll 4
  for (int k = 0; k < HB; ++k) s += y2_f32[(size_t)b * HB + k] * w[HB + k];
  logits[idx] = s;
}

// ---------------------------------------------------------------------------

extern "C" void kernel_launch(void* const* d_in, const int* in_sizes, int n_in,
                              void* d_out, int out_size, void* d_ws, size_t ws_size,
                              hipStream_t stream) {
  (void)in_sizes; (void)n_in; (void)out_size; (void)ws_size;

  const float* x       = (const float*)d_in[0];
  // d_in[1]=hidden1, d_in[2]=hidden2 unused by the reference (carries start at 0)
  const float* snn_fc1 = (const float*)d_in[3];
  const float* snn_fc2 = (const float*)d_in[4];
  const float* rnn_fc1 = (const float*)d_in[5];
  const float* rnn_fc2 = (const float*)d_in[6];
  const float* rnn_fv1 = (const float*)d_in[7];
  const float* rnn_fv2 = (const float*)d_in[8];
  const float* fc_w    = (const float*)d_in[9];
  const float* fc_b    = (const float*)d_in[10];

  // ---- workspace carve (~64 MB) ----
  char* p = (char*)d_ws;
  auto take = [&](size_t bytes) -> char* {
    char* r = p; p += (bytes + 255) & ~(size_t)255; return r;
  };
  const size_t stBf = (size_t)BB * HB * sizeof(__bf16);
  const size_t stF  = (size_t)BB * HB * sizeof(float);

  __bf16* xt   = (__bf16*)take((size_t)TT * BB * 32 * 2);
  __bf16* w_s1 = (__bf16*)take((size_t)32 * 512 * 2);
  __bf16* w_r1 = (__bf16*)take((size_t)32 * 512 * 2);
  __bf16* w_s2 = (__bf16*)take((size_t)1024 * 512 * 2);
  __bf16* w_r2 = (__bf16*)take((size_t)1024 * 512 * 2);
  __bf16* w_v1 = (__bf16*)take((size_t)512 * 512 * 2);
  __bf16* w_v2 = (__bf16*)take((size_t)512 * 512 * 2);
  __bf16* spk1 = (__bf16*)take(stBf);
  __bf16* y1b0 = (__bf16*)take(stBf);
  __bf16* y1b1 = (__bf16*)take(stBf);
  __bf16* y2b0 = (__bf16*)take(stBf);
  __bf16* y2b1 = (__bf16*)take(stBf);
  float*  mem1 = (float*)take(stF);
  float*  mem2 = (float*)take(stF);
  float*  spk2 = (float*)take(stF);
  float*  sum2 = (float*)take(stF);
  __bf16* y1b[2] = { y1b0, y1b1 };
  __bf16* y2b[2] = { y2b0, y2b1 };

  float* out        = (float*)d_out;
  float* out_logits = out;                          // [BB,10]
  float* out_h1y    = out + (size_t)BB * 10;        // [BB,HB]
  float* out_h2y    = out_h1y + (size_t)BB * HB;    // [BB,HB]

  // ---- deterministic state init every launch (capture-safe memsets) ----
  (void)hipMemsetAsync(spk1, 0, stBf, stream);
  (void)hipMemsetAsync(y1b0, 0, stBf, stream);
  (void)hipMemsetAsync(y2b0, 0, stBf, stream);
  (void)hipMemsetAsync(mem1, 0, stF,  stream);
  (void)hipMemsetAsync(mem2, 0, stF,  stream);
  (void)hipMemsetAsync(spk2, 0, stF,  stream);
  (void)hipMemsetAsync(sum2, 0, stF,  stream);

  // ---- pack inputs/weights to bf16 fragment layouts ----
  auto gp = [](size_t n) { return (unsigned)((n + 255) / 256); };
  hnn_pack_x<<<gp((size_t)TT * BB * 32), 256, 0, stream>>>(x, xt);
  hnn_pack_w<<<gp(32   * 512), 256, 0, stream>>>(snn_fc1, w_s1,   28,   32, 512);
  hnn_pack_w<<<gp(32   * 512), 256, 0, stream>>>(rnn_fc1, w_r1,   28,   32, 512);
  hnn_pack_w<<<gp(1024 * 512), 256, 0, stream>>>(snn_fc2, w_s2, 1024, 1024, 512);
  hnn_pack_w<<<gp(1024 * 512), 256, 0, stream>>>(rnn_fc2, w_r2, 1024, 1024, 512);
  hnn_pack_w<<<gp(512  * 512), 256, 0, stream>>>(rnn_fv1, w_v1,  512,  512, 512);
  hnn_pack_w<<<gp(512  * 512), 256, 0, stream>>>(rnn_fv2, w_v2,  512,  512, 512);

  // ---- 28 sequential recurrent steps ----
  for (int t = 0; t < TT; ++t) {
    const int cur = t & 1, nxt = cur ^ 1;
    const int last = (t == TT - 1) ? 1 : 0;
    hnn_layer1<<<256, 256, 0, stream>>>(
        xt + (size_t)t * BB * 32, w_s1, w_r1, w_v1,
        y1b[cur], y1b[nxt], spk1, mem1, out_h1y, last);
    hnn_layer2<<<256, 256, 0, stream>>>(
        spk1, y1b[nxt], w_s2, w_r2, w_v2,
        y2b[cur], y2b[nxt], spk2, mem2, sum2, out_h2y, last);
  }

  // ---- classifier head ----
  hnn_head<<<gp((size_t)BB * 10), 256, 0, stream>>>(sum2, out_h2y, fc_w, fc_b,
                                                    out_logits);
}
